// SparseDecoder_29669634081335
// MI455X (gfx1250) — matrix operand order repaired and checked by reading, compiled, not run on hardware
//
#include <hip/hip_runtime.h>
#include <hip/hip_bf16.h>
#include <math.h>

#define D_MODEL   1024
#define N_HEADS   16
#define HEAD_DIM  64
#define N_LAYERS  2
#define VOCAB     32000
#define SEQ_T     2048
#define HIDDEN    4096
#define TOPK      64

typedef __bf16 bf16_t;
typedef __attribute__((ext_vector_type(16))) __bf16 v16bf;
typedef __attribute__((ext_vector_type(8)))  __bf16 v8bf;
typedef __attribute__((ext_vector_type(8)))  float  v8f;

__device__ __forceinline__ v8f wmma_bf16(v16bf a, v16bf b, v8f c) {
  // D = A(16x32 bf16) x B(32x16 bf16) + C(16x16 f32), wave32
  return __builtin_amdgcn_wmma_f32_16x16x32_bf16(false, a, false, b, (short)0, c, false, false);
}

// ---------------------------------------------------------------------------
// f32 -> bf16 straight conversion (Wemb natural layout == transposed-B layout
// for the tied LM head)
// ---------------------------------------------------------------------------
__global__ __launch_bounds__(256)
void f32_to_bf16_kernel(const float* __restrict__ in, bf16_t* __restrict__ out, size_t n) {
  size_t i = (size_t)blockIdx.x * blockDim.x + threadIdx.x;
  size_t stride = (size_t)gridDim.x * blockDim.x;
  for (; i < n; i += stride) out[i] = (bf16_t)in[i];
}

// ---------------------------------------------------------------------------
// Tiled transpose + convert: in f32 [R][C] -> out bf16 [C][R]  (R,C mult of 32)
// ---------------------------------------------------------------------------
__global__ __launch_bounds__(256)
void transpose_bf16_kernel(const float* __restrict__ in, bf16_t* __restrict__ out,
                           int R, int C) {
  __shared__ float tile[32][33];
  const int c0 = blockIdx.x * 32, r0 = blockIdx.y * 32;
  const int tx = threadIdx.x, ty = threadIdx.y;
#pragma unroll
  for (int j = 0; j < 4; ++j)
    tile[ty + 8 * j][tx] = in[(size_t)(r0 + ty + 8 * j) * C + (c0 + tx)];
  __syncthreads();
#pragma unroll
  for (int j = 0; j < 4; ++j)
    out[(size_t)(c0 + ty + 8 * j) * R + (r0 + tx)] = (bf16_t)tile[tx][ty + 8 * j];
}

// ---------------------------------------------------------------------------
// x[t][d] = Wemb[ids[t]][d] + Wpos[t][d]   (f32 residual stream)
// ---------------------------------------------------------------------------
__global__ __launch_bounds__(256)
void embed_kernel(const int* __restrict__ ids, const float* __restrict__ wemb,
                  const float* __restrict__ wpos, float* __restrict__ x, int D) {
  const int t = blockIdx.x;
  const float* e = wemb + (size_t)ids[t] * D;
  const float* p = wpos + (size_t)t * D;
  float* o = x + (size_t)t * D;
  for (int d = threadIdx.x; d < D; d += 256) o[d] = e[d] + p[d];
}

// ---------------------------------------------------------------------------
// LayerNorm: f32 row in -> bf16 row out (feeds WMMA GEMM A operand)
// ---------------------------------------------------------------------------
__global__ __launch_bounds__(256)
void layernorm_bf16_kernel(const float* __restrict__ x, const float* __restrict__ g,
                           const float* __restrict__ b, bf16_t* __restrict__ out,
                           int cols) {
  const int row = blockIdx.x;
  const float* xr = x + (size_t)row * cols;
  float s1 = 0.f, s2 = 0.f;
  for (int c = threadIdx.x; c < cols; c += 256) { float v = xr[c]; s1 += v; s2 += v * v; }
  __shared__ float sh1[256], sh2[256];
  sh1[threadIdx.x] = s1; sh2[threadIdx.x] = s2;
  __syncthreads();
  for (int o = 128; o > 0; o >>= 1) {
    if (threadIdx.x < o) { sh1[threadIdx.x] += sh1[threadIdx.x + o];
                           sh2[threadIdx.x] += sh2[threadIdx.x + o]; }
    __syncthreads();
  }
  const float mean = sh1[0] / cols;
  const float var  = sh2[0] / cols - mean * mean;
  const float inv  = rsqrtf(var + 1e-5f);
  for (int c = threadIdx.x; c < cols; c += 256)
    out[(size_t)row * cols + c] = (bf16_t)((xr[c] - mean) * inv * g[c] + b[c]);
}

// ---------------------------------------------------------------------------
// Per-(t,head) top-2 |.| feature sparsify (feat_keep = ceil(64*64/2048) = 2),
// packed to head-major bf16: out_rm [H][T][64] and/or out_cm [H][64][T].
// One wave32 per (t,h); 2 elements per lane; top-2 via shfl_xor tree.
// ---------------------------------------------------------------------------
__global__ __launch_bounds__(256)
void sparsify_pack_kernel(const float* __restrict__ x, bf16_t* __restrict__ out_rm,
                          bf16_t* __restrict__ out_cm, int T) {
  const int w    = (int)((blockIdx.x * blockDim.x + threadIdx.x) >> 5);
  const int lane = threadIdx.x & 31;
  const int t = w >> 4;       // / N_HEADS
  const int h = w & 15;
  if (t >= T) return;
  const float* src = x + (size_t)t * D_MODEL + h * HEAD_DIM;
  const float v0 = src[2 * lane], v1 = src[2 * lane + 1];
  const float a0 = fabsf(v0), a1 = fabsf(v1);
  float m1 = fmaxf(a0, a1), m2 = fminf(a0, a1);
  for (int off = 16; off > 0; off >>= 1) {
    float o1 = __shfl_xor(m1, off, 32);
    float o2 = __shfl_xor(m2, off, 32);
    float big = fmaxf(m1, o1);
    float sec = fmaxf(fminf(m1, o1), fmaxf(m2, o2));
    m1 = big; m2 = sec;
  }
  const float thr = m2;
  const float r0 = (a0 >= thr) ? v0 : 0.0f;
  const float r1 = (a1 >= thr) ? v1 : 0.0f;
  if (out_rm) {
    bf16_t* d = out_rm + ((size_t)h * T + t) * HEAD_DIM + 2 * lane;
    d[0] = (bf16_t)r0; d[1] = (bf16_t)r1;
  }
  if (out_cm) {
    out_cm[((size_t)h * HEAD_DIM + 2 * lane    ) * T + t] = (bf16_t)r0;
    out_cm[((size_t)h * HEAD_DIM + 2 * lane + 1) * T + t] = (bf16_t)r1;
  }
}

// ---------------------------------------------------------------------------
// Row-wise top-64 + softmax over causal scores; writes bf16 probabilities.
// One 256-thread block per row; 8 strided elems/thread; iterative block-max
// extraction determines the k-th threshold, then masked softmax.
// ---------------------------------------------------------------------------
__global__ __launch_bounds__(256)
void topk_softmax_kernel(const float* __restrict__ scores, bf16_t* __restrict__ probs,
                         int T, int kkeep) {
  const int row = blockIdx.x;
  const int tid = threadIdx.x;
  const float* s = scores + (size_t)row * T;
  bf16_t* p = probs + (size_t)row * T;
  const float NEG = -__builtin_inff();
  float orig[8], work[8];
#pragma unroll
  for (int i = 0; i < 8; ++i) {
    int c = tid + i * 256;
    float v = (c < T && c <= row) ? s[c] : NEG;
    orig[i] = v; work[i] = v;
  }
  __shared__ float sh[256];
  const int kk = min(kkeep, row + 1);
  float rowmax = NEG, thr = NEG;
  for (int it = 0; it < kk; ++it) {
    float lm = work[0];
#pragma unroll
    for (int i = 1; i < 8; ++i) lm = fmaxf(lm, work[i]);
    sh[tid] = lm;
    __syncthreads();
    for (int o = 128; o > 0; o >>= 1) {
      if (tid < o) sh[tid] = fmaxf(sh[tid], sh[tid + o]);
      __syncthreads();
    }
    const float bm = sh[0];
    __syncthreads();
    if (it == 0) rowmax = bm;
    thr = bm;
#pragma unroll
    for (int i = 0; i < 8; ++i) if (work[i] == bm) work[i] = NEG;
  }
  float e[8];
  float ls = 0.f;
#pragma unroll
  for (int i = 0; i < 8; ++i) {
    float v = orig[i];
    e[i] = (v >= thr) ? expf(v - rowmax) : 0.0f;  // invalid cols have v=-inf -> exp = 0
    ls += e[i];
  }
  sh[tid] = ls;
  __syncthreads();
  for (int o = 128; o > 0; o >>= 1) {
    if (tid < o) sh[tid] += sh[tid + o];
    __syncthreads();
  }
  const float inv = (sh[0] > 0.f) ? (1.0f / sh[0]) : 0.0f;
#pragma unroll
  for (int i = 0; i < 8; ++i) {
    int c = tid + i * 256;
    if (c < T) p[c] = (bf16_t)(e[i] * inv);
  }
}

// ---------------------------------------------------------------------------
// WMMA bf16 GEMM:  C[M][N] = act( scale * A[M][K] x Bt[N][K]^T + bias ) + add
// Bt is B transposed (row n holds column n of B) so every lane's fragment is a
// contiguous 16B/32B load matching the ISA 16x16x32 A/B VGPR layouts.
// 256 thr = 8 waves (4 M-rows x 2 N-cols of waves).
//   NT = # of 16-wide N tiles per wave: NT=4 -> wave tile 32x64, WG tile 128x128
//        (8 WMMA per 2 A + 4 B fragment loads);  NT=2 -> 32x32 / 128x64 for N=64.
// Optional causal mask (-inf above diagonal) with whole-tile early-out, GELU,
// bias, residual add, f32 and/or bf16 outputs, non-temporal streaming stores.
// Prefetches the next K slab into near caches (global_prefetch_b8).
// ---------------------------------------------------------------------------
template <int NT>
__global__ __launch_bounds__(256)
void gemm_bf16_kernel(const bf16_t* __restrict__ A, int lda,
                      const bf16_t* __restrict__ Bt, int ldb,
                      const float* __restrict__ bias,
                      const float* __restrict__ addsrc, int ldadd,
                      float* __restrict__ outF, bf16_t* __restrict__ outB, int ldc,
                      int M, int N, int K,
                      float scale, int causal, int gelu, int streamout) {
  constexpr int BN = NT * 32;            // workgroup tile width
  const int lane = threadIdx.x & 31;
  const int wave = threadIdx.x >> 5;
  const int wr = wave & 3, wc = wave >> 2;
  const int hl = lane >> 4;              // half of wave
  const int r  = lane & 15;
  const int mbase = blockIdx.y * 128 + wr * 32;
  const int nbase = blockIdx.x * BN + wc * (16 * NT);
  (void)M; (void)N;

  if (causal && (int)(blockIdx.x * BN) > (int)(blockIdx.y * 128 + 127)) {
    // whole workgroup tile above the diagonal -> pure -inf, skip the K loop
#pragma unroll
    for (int X = 0; X < 2; ++X)
#pragma unroll
      for (int j = 0; j < NT; ++j) {
        const int nn = nbase + 16 * j + r;
#pragma unroll
        for (int v = 0; v < 8; ++v) {
          const int mm = mbase + 16 * X + 8 * hl + v;
          if (outF) __builtin_nontemporal_store(-__builtin_inff(),
                                                &outF[(size_t)mm * ldc + nn]);
        }
      }
    return;
  }

  v8f acc[2][NT];
#pragma unroll
  for (int X = 0; X < 2; ++X)
#pragma unroll
    for (int j = 0; j < NT; ++j) acc[X][j] = (v8f){};

  const bf16_t* arow0 = A + (size_t)(mbase + r) * lda;
  const bf16_t* arow1 = arow0 + (size_t)16 * lda;
  const bf16_t* brow[NT];
#pragma unroll
  for (int j = 0; j < NT; ++j)
    brow[j] = Bt + (size_t)(nbase + 16 * j + r) * ldb + 8 * hl;

  for (int kb = 0; kb < K; kb += 32) {
    if (kb + 32 < K) {  // gfx1250 global_prefetch_b8, near scope (all levels)
      __builtin_prefetch(arow0 + kb + 32, 0, 3);
      __builtin_prefetch(arow1 + kb + 32, 0, 3);
#pragma unroll
      for (int j = 0; j < NT; ++j) __builtin_prefetch(brow[j] + kb + 32, 0, 3);
    }
    // A fragment: lane {hl,r}: row r, K = kb + 8*hl + [0..7] and kb+16+8*hl+[0..7]
    v8bf a0lo = *(const v8bf*)(arow0 + kb + 8 * hl);
    v8bf a0hi = *(const v8bf*)(arow0 + kb + 16 + 8 * hl);
    v8bf a1lo = *(const v8bf*)(arow1 + kb + 8 * hl);
    v8bf a1hi = *(const v8bf*)(arow1 + kb + 16 + 8 * hl);
    v16bf a0 = __builtin_shufflevector(a0lo, a0hi, 0,1,2,3,4,5,6,7,8,9,10,11,12,13,14,15);
    v16bf a1 = __builtin_shufflevector(a1lo, a1hi, 0,1,2,3,4,5,6,7,8,9,10,11,12,13,14,15);
#pragma unroll
    for (int j = 0; j < NT; ++j) {
      // B fragment: lane {hl,r}: col (nbase+16j+r), K = kb + 8*hl + [0..7] and kb+16+8*hl+[0..7]
      v8bf blo = *(const v8bf*)(brow[j] + kb);
      v8bf bhi = *(const v8bf*)(brow[j] + kb + 16);
      v16bf b = __builtin_shufflevector(blo, bhi, 0,1,2,3,4,5,6,7,8,9,10,11,12,13,14,15);
      acc[0][j] = wmma_bf16(a0, b, acc[0][j]);
      acc[1][j] = wmma_bf16(a1, b, acc[1][j]);
    }
  }

#pragma unroll
  for (int X = 0; X < 2; ++X)
#pragma unroll
    for (int j = 0; j < NT; ++j) {
      const int nn = nbase + 16 * j + r;
#pragma unroll
      for (int v = 0; v < 8; ++v) {
        const int mm = mbase + 16 * X + 8 * hl + v;
        float val = acc[X][j][v] * scale;
        if (bias) val += bias[nn];
        if (gelu) val = 0.5f * val * (1.0f + erff(val * 0.70710678118f));
        if (addsrc) val += addsrc[(size_t)mm * ldadd + nn];
        if (causal && nn > mm) val = -__builtin_inff();
        if (outF) {
          if (streamout) __builtin_nontemporal_store(val, &outF[(size_t)mm * ldc + nn]);
          else outF[(size_t)mm * ldc + nn] = val;
        }
        if (outB) outB[(size_t)mm * ldc + nn] = (bf16_t)val;
      }
    }
}

// ---------------------------------------------------------------------------
// Host orchestration
// ---------------------------------------------------------------------------
extern "C" void kernel_launch(void* const* d_in, const int* in_sizes, int n_in,
                              void* d_out, int out_size, void* d_ws, size_t ws_size,
                              hipStream_t stream) {
  (void)in_sizes; (void)n_in; (void)out_size; (void)ws_size;
  const int*   ids  = (const int*)  d_in[0];
  const float* Wemb = (const float*)d_in[1];
  const float* Wpos = (const float*)d_in[2];
  const float* ln1g = (const float*)d_in[3];
  const float* ln1b = (const float*)d_in[4];
  const float* Wq   = (const float*)d_in[5];
  const float* Wk   = (const float*)d_in[6];
  const float* Wv   = (const float*)d_in[7];
  const float* Wo   = (const float*)d_in[8];
  const float* ln2g = (const float*)d_in[9];
  const float* ln2b = (const float*)d_in[10];
  const float* W1   = (const float*)d_in[11];
  const float* b1   = (const float*)d_in[12];
  const float* W2   = (const float*)d_in[13];
  const float* b2   = (const float*)d_in[14];
  const float* lnfg = (const float*)d_in[15];
  const float* lnfb = (const float*)d_in[16];
  float* logits = (float*)d_out;

  const int T = SEQ_T, D = D_MODEL, HID = HIDDEN, V = VOCAB;
  const size_t DD = (size_t)D * D;

  // deterministic bump allocator over d_ws
  char* wsp = (char*)d_ws;
  auto alloc = [&](size_t bytes) -> char* {
    char* p = wsp;
    wsp += (bytes + 255) & ~(size_t)255;
    return p;
  };
  bf16_t* wembB = (bf16_t*)alloc((size_t)V * D * 2);
  bf16_t* wqT   = (bf16_t*)alloc((size_t)N_LAYERS * DD * 2);
  bf16_t* wkT   = (bf16_t*)alloc((size_t)N_LAYERS * DD * 2);
  bf16_t* wvT   = (bf16_t*)alloc((size_t)N_LAYERS * DD * 2);
  bf16_t* woT   = (bf16_t*)alloc((size_t)N_LAYERS * DD * 2);
  bf16_t* w1T   = (bf16_t*)alloc((size_t)N_LAYERS * HID * D * 2);
  bf16_t* w2T   = (bf16_t*)alloc((size_t)N_LAYERS * D * HID * 2);
  float*  x     = (float*) alloc((size_t)T * D * 4);
  float*  x2    = (float*) alloc((size_t)T * D * 4);
  bf16_t* lnbuf = (bf16_t*)alloc((size_t)T * D * 2);
  float*  qf    = (float*) alloc((size_t)T * D * 4);
  float*  kf    = (float*) alloc((size_t)T * D * 4);
  float*  vf    = (float*) alloc((size_t)T * D * 4);
  bf16_t* qh    = (bf16_t*)alloc((size_t)N_HEADS * T * HEAD_DIM * 2);
  bf16_t* kh    = (bf16_t*)alloc((size_t)N_HEADS * T * HEAD_DIM * 2);
  bf16_t* vt    = (bf16_t*)alloc((size_t)N_HEADS * HEAD_DIM * T * 2);
  float*  scr   = (float*) alloc((size_t)T * T * 4);
  bf16_t* prb   = (bf16_t*)alloc((size_t)T * T * 2);
  bf16_t* attnB = (bf16_t*)alloc((size_t)T * D * 2);
  bf16_t* h1B   = (bf16_t*)alloc((size_t)T * HID * 2);

  auto gemm = [&](const bf16_t* A, int lda, const bf16_t* Bt, int ldb,
                  const float* bias, const float* add, int ldadd,
                  float* outF, bf16_t* outB, int ldc,
                  int M, int N, int K, float scale, int causal, int glu, int so) {
    if (N % 128 == 0) {
      dim3 g(N / 128, M / 128);
      gemm_bf16_kernel<4><<<g, 256, 0, stream>>>(A, lda, Bt, ldb, bias, add, ldadd,
                                                 outF, outB, ldc, M, N, K,
                                                 scale, causal, glu, so);
    } else {
      dim3 g(N / 64, M / 128);
      gemm_bf16_kernel<2><<<g, 256, 0, stream>>>(A, lda, Bt, ldb, bias, add, ldadd,
                                                 outF, outB, ldc, M, N, K,
                                                 scale, causal, glu, so);
    }
  };

  // ---- weight prep: convert / transpose to bf16 once per call ----
  f32_to_bf16_kernel<<<2048, 256, 0, stream>>>(Wemb, wembB, (size_t)V * D);
  for (int l = 0; l < N_LAYERS; ++l) {
    dim3 tb(32, 8);
    dim3 tgDD(D / 32, D / 32);
    transpose_bf16_kernel<<<tgDD, tb, 0, stream>>>(Wq + l * DD, wqT + l * DD, D, D);
    transpose_bf16_kernel<<<tgDD, tb, 0, stream>>>(Wk + l * DD, wkT + l * DD, D, D);
    transpose_bf16_kernel<<<tgDD, tb, 0, stream>>>(Wv + l * DD, wvT + l * DD, D, D);
    transpose_bf16_kernel<<<tgDD, tb, 0, stream>>>(Wo + l * DD, woT + l * DD, D, D);
    dim3 tg1(HID / 32, D / 32);   // W1: [D][HID] -> [HID][D]
    transpose_bf16_kernel<<<tg1, tb, 0, stream>>>(W1 + (size_t)l * D * HID,
                                                  w1T + (size_t)l * HID * D, D, HID);
    dim3 tg2(D / 32, HID / 32);   // W2: [HID][D] -> [D][HID]
    transpose_bf16_kernel<<<tg2, tb, 0, stream>>>(W2 + (size_t)l * HID * D,
                                                  w2T + (size_t)l * D * HID, HID, D);
  }

  // ---- embedding ----
  embed_kernel<<<T, 256, 0, stream>>>(ids, Wemb, Wpos, x, D);

  // ---- transformer layers ----
  for (int l = 0; l < N_LAYERS; ++l) {
    // LN1 -> bf16, then Q/K/V projections (f32 out, sparsified next)
    layernorm_bf16_kernel<<<T, 256, 0, stream>>>(x, ln1g + l * D, ln1b + l * D, lnbuf, D);
    gemm(lnbuf, D, wqT + l * DD, D, nullptr, nullptr, 0, qf, nullptr, D, T, D, D, 1.f, 0, 0, 0);
    gemm(lnbuf, D, wkT + l * DD, D, nullptr, nullptr, 0, kf, nullptr, D, T, D, D, 1.f, 0, 0, 0);
    gemm(lnbuf, D, wvT + l * DD, D, nullptr, nullptr, 0, vf, nullptr, D, T, D, D, 1.f, 0, 0, 0);

    // top-2 feature sparsify + head-major pack (v also transposed for AV GEMM)
    const int spBlocks = (T * N_HEADS) / 8;   // 8 waves per 256-thr block
    sparsify_pack_kernel<<<spBlocks, 256, 0, stream>>>(qf, qh, nullptr, T);
    sparsify_pack_kernel<<<spBlocks, 256, 0, stream>>>(kf, kh, nullptr, T);
    sparsify_pack_kernel<<<spBlocks, 256, 0, stream>>>(vf, nullptr, vt, T);

    // attention, head by head (scores buffer reused)
    for (int h = 0; h < N_HEADS; ++h) {
      // scores = (q @ k^T) / sqrt(hd), causal-masked; streamed NT to keep L2 clean
      gemm(qh + (size_t)h * T * HEAD_DIM, HEAD_DIM,
           kh + (size_t)h * T * HEAD_DIM, HEAD_DIM,
           nullptr, nullptr, 0, scr, nullptr, T,
           T, T, HEAD_DIM, 0.125f, 1, 0, 1);
      topk_softmax_kernel<<<T, 256, 0, stream>>>(scr, prb, T, TOPK);
      // out_h = probs @ v_h   (bf16, written at column slice h*64 of [T][D])
      gemm(prb, T, vt + (size_t)h * HEAD_DIM * T, T,
           nullptr, nullptr, 0, nullptr, attnB + h * HEAD_DIM, D,
           T, HEAD_DIM, T, 1.f, 0, 0, 0);
    }

    // x2 = x + attn_out @ Wo
    gemm(attnB, D, woT + l * DD, D, nullptr, x, D, x2, nullptr, D, T, D, D, 1.f, 0, 0, 0);

    // FFN: x = x2 + gelu(ln2(x2) @ W1 + b1) @ W2 + b2
    layernorm_bf16_kernel<<<T, 256, 0, stream>>>(x2, ln2g + l * D, ln2b + l * D, lnbuf, D);
    gemm(lnbuf, D, w1T + (size_t)l * HID * D, D, b1 + (size_t)l * HID, nullptr, 0,
         nullptr, h1B, HID, T, HID, D, 1.f, 0, 1, 0);
    gemm(h1B, HID, w2T + (size_t)l * D * HID, HID, b2 + (size_t)l * D, x2, D,
         x, nullptr, D, T, D, HID, 1.f, 0, 0, 0);
  }

  // ---- final LN + tied LM head; logits (262 MB) streamed non-temporally ----
  layernorm_bf16_kernel<<<T, 256, 0, stream>>>(x, lnfg, lnfb, lnbuf, D);
  gemm(lnbuf, D, wembB, D, nullptr, nullptr, 0, logits, nullptr, V, T, V, D, 1.f, 0, 0, 1);
}
